// LSTM_60790967108346
// MI455X (gfx1250) — matrix operand order repaired and checked by reading, not compiled, not run on hardware
//
#include <hip/hip_runtime.h>
#include <hip/hip_bf16.h>

// ---------------------------------------------------------------------------
// LSTM for MI455X (gfx1250): bf16 WMMA GEMMs with fp32 accumulation, TDM
// (tensor_load_to_lds) staging with a depth-3 pipeline on TENSORcnt.
// z_t = [x_t | h_{t-1}] @ [Wi;Wh] + b  (K = 1280, N = 4096, M = B = 64)
// Weights stored transposed + gate-interleaved so each 64-col tile holds all
// four gates for 16 hidden units -> single fused kernel per timestep.
// ---------------------------------------------------------------------------

#define BSZ   64
#define TLEN  512
#define DIN   256
#define HID   1024
#define KTOT  1280      // DIN + HID
#define GATES 4096      // 4*HID
#define KCH   32        // K per WMMA chunk (bf16 16x16x32)
#define NCHUNK (KTOT / KCH)   // 40
#define NBUF  4         // LDS pipeline depth-3 + compute buffer

typedef __attribute__((ext_vector_type(16))) __bf16 v16bf;
typedef __attribute__((ext_vector_type(8)))  __bf16 v8bf;
typedef __attribute__((ext_vector_type(8)))  float  v8f;

typedef unsigned int  __attribute__((ext_vector_type(4))) tdm_v4u;
typedef int           __attribute__((ext_vector_type(8))) tdm_v8i;
typedef int           __attribute__((ext_vector_type(4))) tdm_v4i;

union FragAB { v16bf v; v8bf h[2]; };

__device__ __forceinline__ float sigm(float x) {
    return 1.0f / (1.0f + __expf(-x));
}
__device__ __forceinline__ float fast_tanh(float x) {
    x = fminf(fmaxf(x, -15.0f), 15.0f);
    float e = __expf(2.0f * x);
    return (e - 1.0f) / (e + 1.0f);
}

// TDM: DMA a dense 2D tile [64 rows x 32 bf16] global->LDS.
// LDS rows padded 32 -> 40 elements via TDM pad (16 DWORDs data, 4 DWORDs pad)
// so WMMA fragment addressing matches the padded LDS layout.
__device__ __forceinline__ void tdm_load_tile(unsigned int lds_off,
                                              const void* gptr,
                                              unsigned long long stride_elems) {
    unsigned long long ga = (unsigned long long)(uintptr_t)gptr;
    tdm_v4u g0;
    g0[0] = 1u;                                   // count=1 (valid user D#)
    g0[1] = lds_off;                              // lds_addr (bytes)
    g0[2] = (unsigned int)(ga & 0xFFFFFFFFu);     // global_addr[31:0]
    g0[3] = (unsigned int)((ga >> 32) & 0x01FFFFFFu) | (2u << 30); // addr[56:32] | type=2
    tdm_v8i g1;
    g1[0] = (int)((1u << 16)      // data_size = 1 -> 2 bytes
                | (1u << 20)      // pad_enable
                | (3u << 22)      // pad_interval: 2^(3+1)=16 DWORDs (= 32 bf16 row)
                | (3u << 25));    // pad_amount: 3+1 = 4 DWORDs (= 8 bf16)
    g1[1] = (int)(32u << 16);     // tensor_dim0 = 32 (lo16); atomic_barrier_addr = 0
    g1[2] = (int)(64u << 16);     // tensor_dim0 hi = 0 | tensor_dim1 = 64 (lo16)
    g1[3] = (int)(32u << 16);     // tensor_dim1 hi = 0 | tile_dim0 = 32
    g1[4] = 64;                   // tile_dim1 = 64 | tile_dim2 = 0 (unused)
    g1[5] = (int)(stride_elems & 0xFFFFFFFFu);    // tensor_dim0_stride[31:0]
    g1[6] = (int)((stride_elems >> 32) & 0xFFFF); // stride[47:32] | dim1_stride lo = 0
    g1[7] = 0;
    tdm_v4i z4 = {0, 0, 0, 0};            // groups 2/3: tile_dim3/4 = 0 (2D tile)
    tdm_v8i z8 = {0, 0, 0, 0, 0, 0, 0, 0};
    __builtin_amdgcn_tensor_load_to_lds(g0, g1, z4, z4, z8, 0);  // 6-arg form
}

// ---------------- prep kernels ----------------

__global__ void k_cvt_x(const float* __restrict__ x, __bf16* __restrict__ xb, int n) {
    int i = blockIdx.x * blockDim.x + threadIdx.x;
    if (i < n) xb[i] = (__bf16)x[i];
}

// WcT[c][k], c = 4*j + g (gate-interleaved), transposed so WMMA B-fragments
// (column-contiguous K) are unit-stride reads. Also builds permuted bias.
__global__ void k_build_w(const float* __restrict__ Wi, const float* __restrict__ Wh,
                          const float* __restrict__ b,
                          __bf16* __restrict__ WcT, float* __restrict__ bc) {
    int i = blockIdx.x * blockDim.x + threadIdx.x;   // over GATES*KTOT
    if (i >= GATES * KTOT) return;
    int k = i % KTOT;
    int c = i / KTOT;
    int j = c >> 2, g = c & 3;
    int col = g * HID + j;
    float v = (k < DIN) ? Wi[(size_t)k * GATES + col]
                        : Wh[(size_t)(k - DIN) * GATES + col];
    WcT[(size_t)c * KTOT + k] = (__bf16)v;
    if (k == 0) bc[c] = b[col];
}

__global__ void k_zero_state(float* __restrict__ cb, __bf16* __restrict__ hb) {
    int i = blockIdx.x * blockDim.x + threadIdx.x;
    if (i < BSZ * HID) {
        cb[i] = 0.0f;
        __bf16 z = (__bf16)0.0f;
        hb[i] = z;
        hb[i + BSZ * HID] = z;   // ping-pong buffer
    }
}

// ---------------- per-timestep fused GEMM + gates ----------------

__launch_bounds__(128, 1)
__attribute__((amdgpu_waves_per_eu(1)))
__global__ void k_lstm_step(const __bf16* __restrict__ xb,
                            const __bf16* __restrict__ WcT,
                            const float*  __restrict__ bc,
                            const __bf16* __restrict__ hin,
                            __bf16*       __restrict__ hout,
                            float*        __restrict__ cbuf,
                            float*        __restrict__ out,
                            int t) {
    __shared__ __align__(16) __bf16 As[NBUF][64][40];  // A chunk: [row][k], pad 32->40
    __shared__ __align__(16) __bf16 Bs[NBUF][64][40];  // B^T chunk: [n][k]
    __shared__ float Zs[64][68];                       // z tile, pad 64->68

    const int tid  = threadIdx.x;
    const int lane = tid & 31;
    const int wave = tid >> 5;
    const int nb   = blockIdx.x * 64;                  // column base (interleaved space)

    v8f acc[4] = {};

    // Issue TDM loads for chunk index c into buffer c&3 (wave 0 only; EXEC
    // is ignored by TDM, so guard at wave granularity to issue exactly once).
    auto issue_chunk = [&](int c) {
        const int kc  = c * KCH;
        const int buf = c & (NBUF - 1);
        const __bf16* asrc;
        unsigned long long astride;
        if (kc < DIN) { asrc = xb + (size_t)t * DIN + kc;  astride = (size_t)TLEN * DIN; }
        else          { asrc = hin + (kc - DIN);           astride = HID; }
        tdm_load_tile((unsigned int)(uintptr_t)&As[buf][0][0], asrc, astride);
        tdm_load_tile((unsigned int)(uintptr_t)&Bs[buf][0][0],
                      WcT + (size_t)nb * KTOT + kc, KTOT);
    };

    if (wave == 0) {           // prologue: chunks 0..2 in flight (6 tensor ops)
        issue_chunk(0);
        issue_chunk(1);
        issue_chunk(2);
    }

    // A fragment (16x32 bf16): lanes 0-15 row M=lane, K {0..7,16..23};
    // lanes 16-31 same rows, K {8..15,24..31} -> two contiguous 16B LDS reads.
    const int mrow  = wave * 16 + (lane & 15);
    const int khalf = (lane >> 4) * 8;
    // B fragment (32x16): lane holds column n = lane%16, K {0..7,16..23} or {8..15,24..31}.
    const int kq    = (lane >> 4) * 16;

    for (int c = 0; c < NCHUNK; ++c) {
        const int buf = c & (NBUF - 1);
        if (wave == 0) {
            // In-order TDM completion: <=4 outstanding ops => chunks <= c landed.
            if (c + 2 < NCHUNK) __builtin_amdgcn_s_wait_tensorcnt(4);
            else                __builtin_amdgcn_s_wait_tensorcnt(0);
        }
        __syncthreads();   // chunk c visible to all waves; chunk c-1 reads done
        if (wave == 0 && c + 3 < NCHUNK)
            issue_chunk(c + 3);   // overwrites buffer (c-1)&3: safe past barrier

        // Issue ALL fragment loads (10x ds_load_b128) before any WMMA so the
        // scheduler can overlap LDS latency with WMMAs via partial dscnt waits.
        FragAB a, bfr[4];
        a.h[0] = *(const v8bf*)&As[buf][mrow][khalf];
        a.h[1] = *(const v8bf*)&As[buf][mrow][khalf + 16];
#pragma unroll
        for (int nt = 0; nt < 4; ++nt) {
            const int coln = nt * 16 + (lane & 15);
            bfr[nt].h[0] = *(const v8bf*)&Bs[buf][coln][khalf];
            bfr[nt].h[1] = *(const v8bf*)&Bs[buf][coln][khalf + 16];
        }
#pragma unroll
        for (int nt = 0; nt < 4; ++nt)
            acc[nt] = __builtin_amdgcn_wmma_f32_16x16x32_bf16(
                false, a.v, false, bfr[nt].v, (short)0, acc[nt], false, false);
    }

    // Stage z tile to LDS: acc element r of lane L is (M = r + 8*(L>=16), N = L%16).
    __syncthreads();
#pragma unroll
    for (int nt = 0; nt < 4; ++nt)
#pragma unroll
        for (int r = 0; r < 8; ++r) {
            int row = wave * 16 + r + ((lane >> 4) << 3);
            Zs[row][nt * 16 + (lane & 15)] = acc[nt][r];
        }
    __syncthreads();

    // Gate fusion: columns 4u+{0,1,2,3} = {i,f,g,o} for hidden unit j.
    for (int item = tid; item < 64 * 16; item += 128) {
        int r = item >> 4;
        int u = item & 15;
        int j = (nb >> 2) + u;
        float iv = Zs[r][4 * u + 0] + bc[nb + 4 * u + 0];
        float fv = Zs[r][4 * u + 1] + bc[nb + 4 * u + 1];
        float gv = Zs[r][4 * u + 2] + bc[nb + 4 * u + 2];
        float ov = Zs[r][4 * u + 3] + bc[nb + 4 * u + 3];
        float cold = cbuf[r * HID + j];
        float cn = sigm(fv) * cold + sigm(iv) * fast_tanh(gv);
        float hn = sigm(ov) * fast_tanh(cn);
        cbuf[r * HID + j] = cn;
        hout[r * HID + j] = (__bf16)hn;
        out[((size_t)r * TLEN + t) * HID + j] = hn;
    }
}

// ---------------- host ----------------

extern "C" void kernel_launch(void* const* d_in, const int* in_sizes, int n_in,
                              void* d_out, int out_size, void* d_ws, size_t ws_size,
                              hipStream_t stream) {
    const float* x  = (const float*)d_in[0];
    const float* Wi = (const float*)d_in[1];
    const float* Wh = (const float*)d_in[2];
    const float* b  = (const float*)d_in[3];
    float* out = (float*)d_out;
    (void)in_sizes; (void)n_in; (void)out_size; (void)ws_size;

    char* ws = (char*)d_ws;
    const size_t sz_wct = (size_t)GATES * KTOT * sizeof(__bf16);     // 10,485,760
    const size_t sz_xb  = (size_t)BSZ * TLEN * DIN * sizeof(__bf16); // 16,777,216
    const size_t sz_bc  = (size_t)GATES * sizeof(float);             // 16,384
    const size_t sz_cb  = (size_t)BSZ * HID * sizeof(float);         // 262,144

    __bf16* WcT = (__bf16*)(ws);
    __bf16* xb  = (__bf16*)(ws + sz_wct);
    float*  bc  = (float*)(ws + sz_wct + sz_xb);
    float*  cb  = (float*)(ws + sz_wct + sz_xb + sz_bc);
    __bf16* hb  = (__bf16*)(ws + sz_wct + sz_xb + sz_bc + sz_cb);    // 2 x (64x1024)

    const int nx = BSZ * TLEN * DIN;
    k_cvt_x<<<(nx + 255) / 256, 256, 0, stream>>>(x, xb, nx);

    const int nw = GATES * KTOT;
    k_build_w<<<(nw + 255) / 256, 256, 0, stream>>>(Wi, Wh, b, WcT, bc);

    k_zero_state<<<(BSZ * HID + 255) / 256, 256, 0, stream>>>(cb, hb);

    for (int t = 0; t < TLEN; ++t) {
        const __bf16* hin = hb + (size_t)(t & 1) * BSZ * HID;
        __bf16* hout      = hb + (size_t)((t + 1) & 1) * BSZ * HID;
        k_lstm_step<<<GATES / 64, 128, 0, stream>>>(xb, WcT, bc, hin, hout, cb, out, t);
    }
}
